// OptimizedMambaBlock_31662498906797
// MI455X (gfx1250) — matrix operand order repaired and checked by reading, not compiled, not run on hardware
//
#include <hip/hip_runtime.h>
#include <hip/hip_bf16.h>
#include <math.h>

typedef __attribute__((ext_vector_type(16))) _Float16 v16h;
typedef __attribute__((ext_vector_type(8)))  _Float16 v8h;
typedef __attribute__((ext_vector_type(8)))  float    v8f;

#define DIM_C   256
#define DSTATE  16
#define DINNER  512
#define DTRANK  16
#define NBATCH  8
#define SEQLEN  1024
#define MTOT    (NBATCH*SEQLEN)   // 8192 tokens

#define BM 128
#define BN 128
#define BK 32
#define LDS_STRIDE (BK+8)         // pad to kill LDS bank conflicts

#define USE_ASYNC_LDS 1

union V16 { v16h v; v8h h[2]; };

// ---------------------------------------------------------------------------
// CDNA5 async global->LDS copy (ASYNCcnt path). Per-lane 16B transfer:
// VDST = LDS byte address (low 32 bits of generic shared pointer),
// VADDR = 64-bit global address, GV mode.
// ---------------------------------------------------------------------------
__device__ __forceinline__ void async_copy_b128(_Float16* dst_lds,
                                                const _Float16* src_global) {
    unsigned l = (unsigned)(uintptr_t)dst_lds;   // LDS offset = addr[31:0]
    asm volatile("global_load_async_to_lds_b128 %0, %1, off"
                 :: "v"(l), "v"(src_global) : "memory");
}
__device__ __forceinline__ void async_wait0() {
    asm volatile("s_wait_asynccnt 0x0" ::: "memory");
}

// ---------------------------------------------------------------------------
// f32 -> f16 weight conversion
// ---------------------------------------------------------------------------
__global__ void convert_f32_f16(const float* __restrict__ in,
                                _Float16* __restrict__ out, int n) {
    int i = blockIdx.x * 256 + threadIdx.x;
    if (i < n) out[i] = (_Float16)in[i];
}

// ---------------------------------------------------------------------------
// LayerNorm: one wave32 per 256-wide row, 8 rows per 256-thread block.
// ---------------------------------------------------------------------------
template <bool TIN, bool WF32>
__global__ __launch_bounds__(256)
void layernorm_kernel(const float* __restrict__ in,
                      const float* __restrict__ g,
                      const float* __restrict__ bb,
                      _Float16* __restrict__ out16,
                      float* __restrict__ rawf)
{
    const int wave = threadIdx.x >> 5;
    const int lane = threadIdx.x & 31;
    const int row  = blockIdx.x * 8 + wave;

    float v[8];
    #pragma unroll
    for (int j = 0; j < 8; ++j) {
        int c = lane + j * 32;
        size_t idx = TIN ? ((size_t)(row >> 10) * DIM_C + c) * SEQLEN + (row & 1023)
                         : (size_t)row * DIM_C + c;
        v[j] = in[idx];
    }
    float s = 0.f;
    #pragma unroll
    for (int j = 0; j < 8; ++j) s += v[j];
    #pragma unroll
    for (int m = 16; m >= 1; m >>= 1) s += __shfl_xor(s, m, 32);
    const float mean = s * (1.f / 256.f);

    float vs = 0.f;
    #pragma unroll
    for (int j = 0; j < 8; ++j) { float d = v[j] - mean; vs += d * d; }
    #pragma unroll
    for (int m = 16; m >= 1; m >>= 1) vs += __shfl_xor(vs, m, 32);
    const float rstd = rsqrtf(vs * (1.f / 256.f) + 1e-5f);

    #pragma unroll
    for (int j = 0; j < 8; ++j) {
        int c = lane + j * 32;
        float o = (v[j] - mean) * rstd * g[c] + bb[c];
        out16[(size_t)row * DIM_C + c] = (_Float16)o;
        if (WF32) rawf[(size_t)row * DIM_C + c] = v[j];
    }
}

// ---------------------------------------------------------------------------
// Tile loader with compile-time guards.
// ROWG: row bound check needed (N % BN != 0 for the W matrix)
// KG:   K bound check needed (K % BK != 0)
// ASYNC: use global_load_async_to_lds_b128 (only legal when no guards)
// 256 threads: 2 threads per row, 16 f16 (2 x b128) each.
// ---------------------------------------------------------------------------
template <bool ROWG, bool KG, bool ASYNC>
__device__ __forceinline__ void load_tile(
    const _Float16* __restrict__ src, int ld, int rows_total,
    int row0, int k0, int K, _Float16* __restrict__ dst)
{
    const int t    = threadIdx.x;
    const int r    = t >> 1;
    const int ks   = (t & 1) << 4;
    const int grow = row0 + r;
    _Float16* drow = dst + r * LDS_STRIDE + ks;
    const _Float16* srow = src + (size_t)grow * ld + k0 + ks;

    if constexpr (!ROWG && !KG) {
        if constexpr (ASYNC) {
            async_copy_b128(drow, srow);
            async_copy_b128(drow + 8, srow + 8);
        } else {
            const v8h* p = (const v8h*)srow;
            *(v8h*)(drow)     = p[0];
            *(v8h*)(drow + 8) = p[1];
        }
    } else {
        const bool row_ok = !ROWG || (grow < rows_total);
        const bool k_full = !KG   || (k0 + ks + 16 <= K);
        if (row_ok && k_full) {
            const v8h* p = (const v8h*)srow;
            *(v8h*)(drow)     = p[0];
            *(v8h*)(drow + 8) = p[1];
        } else if (!row_ok || (KG && (k0 + ks >= K))) {
            v8h z = {};
            *(v8h*)(drow)     = z;
            *(v8h*)(drow + 8) = z;
        } else {
            #pragma unroll
            for (int i = 0; i < 16; ++i) {
                int kk = k0 + ks + i;
                drow[i] = (kk < K) ? src[(size_t)grow * ld + kk] : (_Float16)0.0f;
            }
        }
    }
}

// ---------------------------------------------------------------------------
// WMMA GEMM: out = epilogue(A(MxK) x W(NxK)^T), M = 8192 (grid-exact).
// 8 waves; wave (wm,wn) owns 32x64 via 2x4 v_wmma_f32_16x16x32_f16.
// EPI 0: store f16            EPI 2: +bias, softplus -> f32
// EPI 3: +res -> f32          EPI 4: +bias, gelu -> f16
// EPI 5: +bias +res, store f32 transposed to (B, C, HW)
// ---------------------------------------------------------------------------
template <int EPI, int N, int K>
__global__ __launch_bounds__(256)
void wmma_gemm_kernel(const _Float16* __restrict__ A, int lda,
                      const _Float16* __restrict__ W, int ldw,
                      const float* __restrict__ bias,
                      const float* __restrict__ res,
                      _Float16* __restrict__ out16,
                      float*    __restrict__ out32,
                      float*    __restrict__ outT,
                      int ldo)
{
    constexpr bool KG = (K % BK) != 0;
    constexpr bool NG = (N % BN) != 0;
    constexpr bool A_ASYNC = USE_ASYNC_LDS && !KG;
    constexpr bool W_ASYNC = USE_ASYNC_LDS && !KG && !NG;

    __shared__ __align__(16) _Float16 sA[BM * LDS_STRIDE];
    __shared__ __align__(16) _Float16 sB[BN * LDS_STRIDE];

    const int wave = threadIdx.x >> 5;
    const int wm   = wave >> 1;        // 0..3 -> 32 rows each
    const int wn   = wave & 1;         // 0..1 -> 64 cols each
    const int lane = threadIdx.x & 31;
    const int lh   = lane & 15;
    const int hi   = lane >> 4;

    const int m0 = blockIdx.y * BM;
    const int n0 = blockIdx.x * BN;

    v8f acc[2][4];
    #pragma unroll
    for (int am = 0; am < 2; ++am) {
        #pragma unroll
        for (int bn = 0; bn < 4; ++bn) { v8f z = {}; acc[am][bn] = z; }
    }

    for (int k0 = 0; k0 < K; k0 += BK) {
        load_tile<false, KG, A_ASYNC>(A, lda, MTOT, m0, k0, K, sA);
        load_tile<NG,    KG, W_ASYNC>(W, ldw, N,    n0, k0, K, sB);
        if (k0 + BK < K) {  // global_prefetch_b8 of next K-tile
            __builtin_prefetch(A + (size_t)(m0 + (threadIdx.x >> 1)) * lda + k0 + BK, 0, 1);
            __builtin_prefetch(W + (size_t)(n0 + (threadIdx.x >> 1)) * ldw + k0 + BK, 0, 1);
        }
        if constexpr (A_ASYNC || W_ASYNC) async_wait0();
        __syncthreads();

        // A frag (16x32): lanes 0-15 K{0-7,16-23}, lanes 16-31 K{8-15,24-31}
        V16 af[2];
        #pragma unroll
        for (int am = 0; am < 2; ++am) {
            const _Float16* pa = sA + (wm * 32 + am * 16 + lh) * LDS_STRIDE + hi * 8;
            af[am].h[0] = *(const v8h*)(pa);
            af[am].h[1] = *(const v8h*)(pa + 16);
        }
        // B frag (32x16): lanes 0-15 K{0-7,16-23}, lanes 16-31 K{8-15,24-31}
        V16 bf[4];
        #pragma unroll
        for (int bn = 0; bn < 4; ++bn) {
            const _Float16* pb = sB + (wn * 64 + bn * 16 + lh) * LDS_STRIDE + hi * 8;
            bf[bn].h[0] = *(const v8h*)(pb);
            bf[bn].h[1] = *(const v8h*)(pb + 16);
        }
        #pragma unroll
        for (int am = 0; am < 2; ++am) {
            #pragma unroll
            for (int bn = 0; bn < 4; ++bn) {
                acc[am][bn] = __builtin_amdgcn_wmma_f32_16x16x32_f16(
                    false, af[am].v, false, bf[bn].v, (short)0, acc[am][bn], false, false);
            }
        }
        __syncthreads();
    }

    // D layout: VGPR r -> M = r + 8*hi, N = lh (per 16x16 tile)
    #pragma unroll
    for (int am = 0; am < 2; ++am) {
        #pragma unroll
        for (int bn = 0; bn < 4; ++bn) {
            const int c = n0 + wn * 64 + bn * 16 + lh;
            if (NG && c >= N) continue;
            const int rbase = m0 + wm * 32 + am * 16 + hi * 8;
            #pragma unroll
            for (int r = 0; r < 8; ++r) {
                const int rr = rbase + r;
                float v = acc[am][bn][r];
                if (EPI == 0) {
                    out16[(size_t)rr * ldo + c] = (_Float16)v;
                } else if (EPI == 2) {
                    v += bias[c];
                    v = (v > 20.f) ? v : log1pf(__expf(v));
                    out32[(size_t)rr * ldo + c] = v;
                } else if (EPI == 3) {
                    v += res[(size_t)rr * ldo + c];
                    out32[(size_t)rr * ldo + c] = v;
                } else if (EPI == 4) {
                    v += bias[c];
                    v = 0.5f * v * (1.f + erff(v * 0.70710678118f));
                    out16[(size_t)rr * ldo + c] = (_Float16)v;
                } else if (EPI == 5) {
                    v += bias[c] + res[(size_t)rr * DIM_C + c];
                    const int b  = rr >> 10;
                    const int hw = rr & 1023;
                    outT[((size_t)b * DIM_C + c) * SEQLEN + hw] = v;
                }
            }
        }
    }
}

// ---------------------------------------------------------------------------
// Depthwise causal conv (k=4) + SiLU. idx = m*512 + d.
// ---------------------------------------------------------------------------
__global__ __launch_bounds__(256)
void conv_silu_kernel(const _Float16* __restrict__ xz16,
                      const float* __restrict__ conv_w,
                      const float* __restrict__ conv_b,
                      float* __restrict__ xc32,
                      _Float16* __restrict__ xc16)
{
    const int idx = blockIdx.x * 256 + threadIdx.x;   // < MTOT*DINNER
    const int d   = idx & (DINNER - 1);
    const int m   = idx >> 9;
    const int l   = m & (SEQLEN - 1);
    const int b   = m >> 10;
    float acc = conv_b[d];
    #pragma unroll
    for (int j = 0; j < 4; ++j) {
        int ls = l - 3 + j;
        if (ls >= 0)
            acc += conv_w[d * 4 + j] *
                   (float)xz16[((size_t)(b << 10) + ls) * (2 * DINNER) + d];
    }
    float s = acc / (1.f + __expf(-acc));   // SiLU
    xc32[idx] = s;
    xc16[idx] = (_Float16)s;
}

// ---------------------------------------------------------------------------
// Selective scan: one thread per (batch, channel); 16-state in VGPRs.
// dbl16 row = [dt_r(16) | B(16) | C(16)] f16.
// ---------------------------------------------------------------------------
__global__ __launch_bounds__(512)
void scan_kernel(const float* __restrict__ dt,
                 const float* __restrict__ u,
                 const _Float16* __restrict__ dbl16,
                 const float* __restrict__ A_log,
                 const float* __restrict__ Dp,
                 float* __restrict__ y)
{
    const int d = threadIdx.x;
    const int b = blockIdx.x;
    float Av[DSTATE], h[DSTATE];
    #pragma unroll
    for (int s = 0; s < DSTATE; ++s) {
        Av[s] = -__expf(A_log[d * DSTATE + s]);
        h[s]  = 0.f;
    }
    const float Dpv = Dp[d];
    for (int l = 0; l < SEQLEN; ++l) {
        const size_t m   = (size_t)b * SEQLEN + l;
        const float dtv  = dt[m * DINNER + d];
        const float uv   = u[m * DINNER + d];
        const _Float16* bc = dbl16 + m * 48 + DTRANK;
        float ys = 0.f;
        #pragma unroll
        for (int s = 0; s < DSTATE; ++s) {
            const float Bs = (float)bc[s];
            const float Cs = (float)bc[DSTATE + s];
            h[s] = __expf(dtv * Av[s]) * h[s] + dtv * Bs * uv;
            ys  += h[s] * Cs;
        }
        y[m * DINNER + d] = ys + uv * Dpv;
    }
}

// ---------------------------------------------------------------------------
// y * silu(z) -> f16
// ---------------------------------------------------------------------------
__global__ __launch_bounds__(256)
void gate_kernel(const float* __restrict__ y,
                 const _Float16* __restrict__ xz16,
                 _Float16* __restrict__ yz16)
{
    const int idx = blockIdx.x * 256 + threadIdx.x;   // < MTOT*DINNER
    const int d   = idx & (DINNER - 1);
    const int m   = idx >> 9;
    float z = (float)xz16[(size_t)m * (2 * DINNER) + DINNER + d];
    float s = z / (1.f + __expf(-z));
    yz16[idx] = (_Float16)(y[idx] * s);
}

// ---------------------------------------------------------------------------
extern "C" void kernel_launch(void* const* d_in, const int* in_sizes, int n_in,
                              void* d_out, int out_size, void* d_ws, size_t ws_size,
                              hipStream_t stream)
{
    (void)in_sizes; (void)n_in; (void)out_size; (void)ws_size;
    const float* x          = (const float*)d_in[0];
    const float* ln_g       = (const float*)d_in[1];
    const float* ln_b       = (const float*)d_in[2];
    const float* in_proj_w  = (const float*)d_in[3];
    const float* conv_w     = (const float*)d_in[4];
    const float* conv_b     = (const float*)d_in[5];
    const float* x_proj_w   = (const float*)d_in[6];
    const float* dt_proj_w  = (const float*)d_in[7];
    const float* dt_proj_b  = (const float*)d_in[8];
    const float* A_log      = (const float*)d_in[9];
    const float* Dp         = (const float*)d_in[10];
    const float* out_proj_w = (const float*)d_in[11];
    const float* mlp_ln_g   = (const float*)d_in[12];
    const float* mlp_ln_b   = (const float*)d_in[13];
    const float* mlp_w1     = (const float*)d_in[14];
    const float* mlp_b1     = (const float*)d_in[15];
    const float* mlp_w2     = (const float*)d_in[16];
    const float* mlp_b2     = (const float*)d_in[17];

    char* ws = (char*)d_ws;
    size_t off = 0;
    auto alloc = [&](size_t bytes) -> char* {
        char* p = ws + off;
        off += (bytes + 255) & ~(size_t)255;
        return p;
    };

    _Float16* w_in16  = (_Float16*)alloc((size_t)2 * DINNER * DIM_C * 2);
    _Float16* w_xp16  = (_Float16*)alloc((size_t)48 * DINNER * 2);
    _Float16* w_dt16  = (_Float16*)alloc((size_t)DINNER * DTRANK * 2);
    _Float16* w_out16 = (_Float16*)alloc((size_t)DIM_C * DINNER * 2);
    _Float16* w_m1_16 = (_Float16*)alloc((size_t)4 * DIM_C * DIM_C * 2);
    _Float16* w_m2_16 = (_Float16*)alloc((size_t)DIM_C * 4 * DIM_C * 2);
    _Float16* xn16    = (_Float16*)alloc((size_t)MTOT * DIM_C * 2);
    _Float16* xz16    = (_Float16*)alloc((size_t)MTOT * 2 * DINNER * 2);
    float*    xc32    = (float*)   alloc((size_t)MTOT * DINNER * 4);
    _Float16* xc16    = (_Float16*)alloc((size_t)MTOT * DINNER * 2);
    _Float16* dbl16   = (_Float16*)alloc((size_t)MTOT * 48 * 2);
    float*    dt32    = (float*)   alloc((size_t)MTOT * DINNER * 4);
    float*    y32     = (float*)   alloc((size_t)MTOT * DINNER * 4);
    _Float16* yz16    = (_Float16*)alloc((size_t)MTOT * DINNER * 2);
    float*    xf32    = (float*)   alloc((size_t)MTOT * DIM_C * 4);
    _Float16* hn16    = (_Float16*)alloc((size_t)MTOT * DIM_C * 2);
    _Float16* h1_16   = (_Float16*)alloc((size_t)MTOT * 4 * DIM_C * 2);

    auto cvt = [&](const float* src, _Float16* dst, int n) {
        convert_f32_f16<<<(n + 255) / 256, 256, 0, stream>>>(src, dst, n);
    };
    cvt(in_proj_w,  w_in16,  2 * DINNER * DIM_C);
    cvt(x_proj_w,   w_xp16,  48 * DINNER);
    cvt(dt_proj_w,  w_dt16,  DINNER * DTRANK);
    cvt(out_proj_w, w_out16, DIM_C * DINNER);
    cvt(mlp_w1,     w_m1_16, 4 * DIM_C * DIM_C);
    cvt(mlp_w2,     w_m2_16, DIM_C * 4 * DIM_C);

    const dim3 blk(256);

    // LN1 over raw x (NCHW) -> xn16 (f16) + xf32 residual base
    layernorm_kernel<true, true><<<MTOT / 8, blk, 0, stream>>>(x, ln_g, ln_b, xn16, xf32);

    // in_proj: (8192x256) x (1024x256)^T -> xz16
    wmma_gemm_kernel<0, 1024, 256><<<dim3(8, 64), blk, 0, stream>>>(
        xn16, DIM_C, w_in16, DIM_C,
        nullptr, nullptr, xz16, nullptr, nullptr, 2 * DINNER);

    // depthwise conv + SiLU -> xc32 / xc16
    conv_silu_kernel<<<(MTOT * DINNER) / 256, blk, 0, stream>>>(xz16, conv_w, conv_b, xc32, xc16);

    // x_proj: (8192x512) x (48x512)^T -> dbl16 (dt_r | B | C)
    wmma_gemm_kernel<0, 48, 512><<<dim3(1, 64), blk, 0, stream>>>(
        xc16, DINNER, w_xp16, DINNER,
        nullptr, nullptr, dbl16, nullptr, nullptr, 48);

    // dt_proj + softplus: (8192x16) x (512x16)^T -> dt32
    wmma_gemm_kernel<2, 512, 16><<<dim3(4, 64), blk, 0, stream>>>(
        dbl16, 48, w_dt16, DTRANK,
        dt_proj_b, nullptr, nullptr, dt32, nullptr, DINNER);

    // selective scan -> y32
    scan_kernel<<<NBATCH, 512, 0, stream>>>(dt32, xc32, dbl16, A_log, Dp, y32);

    // gating -> yz16
    gate_kernel<<<(MTOT * DINNER) / 256, blk, 0, stream>>>(y32, xz16, yz16);

    // out_proj + residual: (8192x512) x (256x512)^T + xf -> xf32
    wmma_gemm_kernel<3, 256, 512><<<dim3(2, 64), blk, 0, stream>>>(
        yz16, DINNER, w_out16, DINNER,
        nullptr, xf32, nullptr, xf32, nullptr, DIM_C);

    // LN2 -> hn16
    layernorm_kernel<false, false><<<MTOT / 8, blk, 0, stream>>>(
        xf32, mlp_ln_g, mlp_ln_b, hn16, nullptr);

    // MLP fc1 + bias + GELU: (8192x256) x (1024x256)^T -> h1_16
    wmma_gemm_kernel<4, 1024, 256><<<dim3(8, 64), blk, 0, stream>>>(
        hn16, DIM_C, w_m1_16, DIM_C,
        mlp_b1, nullptr, h1_16, nullptr, nullptr, 4 * DIM_C);

    // MLP fc2 + bias + residual, transposed store to (B, C, HW)
    wmma_gemm_kernel<5, 256, 1024><<<dim3(2, 64), blk, 0, stream>>>(
        h1_16, 4 * DIM_C, w_m2_16, 4 * DIM_C,
        mlp_b2, xf32, nullptr, nullptr, (float*)d_out, DIM_C);
}
